// VectorQuantize_12902081757240
// MI455X (gfx1250) — matrix operand order repaired and checked by reading, not compiled, not run on hardware
//
#include <hip/hip_runtime.h>

// VQ quantization for MI455X (gfx1250, wave32, WMMA).
// dist(n,k) = ||x_n||^2 - 2 x_n.c_k + ||c_k||^2 ; argmin over k; gather; commit loss.
// f16 WMMA (v_wmma_f32_16x16x32_f16) for the 65536x1024x256 distance matmul;
// codebook staged into LDS via CDNA5 async global->LDS loads when available.

#define D_DIM        256
#define K_CODES      1024
#define N_ROWS       (16 * 4096)          // B*T
#define ROWS_PER_BLK 128                  // 8 waves x 16 rows
#define WAVES_PER_BLK 8
#define CHUNK_CODES  64                   // codebook codes staged in LDS per chunk
#define LDS_ROW_H    264                  // 256 halves + 8 pad (528B row, 16B aligned, bank-friendly)

typedef __attribute__((ext_vector_type(16))) _Float16 v16h;
typedef __attribute__((ext_vector_type(8)))  _Float16 v8h;
typedef __attribute__((ext_vector_type(8)))  float    v8f;
typedef __attribute__((ext_vector_type(4)))  int      v4i;

#define AS1 __attribute__((address_space(1)))
#define AS3 __attribute__((address_space(3)))

#if defined(__has_builtin)
#if __has_builtin(__builtin_amdgcn_global_load_async_to_lds_b128) && \
    __has_builtin(__builtin_amdgcn_s_wait_asynccnt)
#define USE_ASYNC_LDS 1
#endif
#endif

// ---------------- Prep: codebook f32 -> f16, plus ||c||^2 per code -------------
__global__ __launch_bounds__(256) void vq_prep_codebook(const float* __restrict__ cb,
                                                        _Float16* __restrict__ cb_h,
                                                        float* __restrict__ c_sq) {
    const int lane = threadIdx.x & 31;
    const int wave = threadIdx.x >> 5;
    const int row  = blockIdx.x * 8 + wave;              // one wave per codebook row
    const float* p = cb + row * D_DIM + lane * 8;
    float4 f0 = *(const float4*)p;
    float4 f1 = *(const float4*)(p + 4);
    float t[8] = {f0.x, f0.y, f0.z, f0.w, f1.x, f1.y, f1.z, f1.w};
    v8h hv;
    float ss = 0.f;
#pragma unroll
    for (int i = 0; i < 8; ++i) { hv[i] = (_Float16)t[i]; ss += t[i] * t[i]; }
    *(v8h*)(cb_h + row * D_DIM + lane * 8) = hv;
#pragma unroll
    for (int off = 16; off >= 1; off >>= 1) ss += __shfl_xor(ss, off, 32);
    if (lane == 0) c_sq[row] = ss;
}

// ---------------- Main: WMMA distance + argmin + gather + loss partials --------
__global__ __launch_bounds__(256) void vq_main(const float* __restrict__ x,
                                               const float* __restrict__ cb,
                                               const _Float16* __restrict__ cb_h,
                                               const float* __restrict__ c_sq,
                                               float* __restrict__ out,
                                               float* __restrict__ idx_out,
                                               float* __restrict__ partials) {
    __shared__ _Float16 lds_cb[CHUNK_CODES * LDS_ROW_H];  // ~33.8 KB
    __shared__ float    lds_csq[K_CODES];                 // 4 KB

    const int tid  = threadIdx.x;
    const int lane = tid & 31;
    const int wave = tid >> 5;
    const int hlf  = lane >> 4;           // 0: lanes 0-15, 1: lanes 16-31
    const int l15  = lane & 15;
    const int rowBase = blockIdx.x * ROWS_PER_BLK + wave * 16;

    // Stage ||c||^2 once.
#pragma unroll
    for (int i = 0; i < 4; ++i) lds_csq[tid + i * 256] = c_sq[tid + i * 256];

    // Build resident A fragments (16x32 f16 per k-slice, 8 slices = full D) and x_sq.
    // A layout: lanes 0-15 hold row l15, K in {0..7, 16..23} of each 32-slice;
    //           lanes 16-31 hold K in {8..15, 24..31}.
    const float* xrow = x + (size_t)(rowBase + l15) * D_DIM;
    v16h a[8];
    float sumsq = 0.f;
#pragma unroll
    for (int s = 0; s < 8; ++s) {
        const float* p = xrow + s * 32 + hlf * 8;
        float4 f0 = *(const float4*)p;
        float4 f1 = *(const float4*)(p + 4);
        float4 f2 = *(const float4*)(p + 16);
        float4 f3 = *(const float4*)(p + 20);
        float t[16] = {f0.x, f0.y, f0.z, f0.w, f1.x, f1.y, f1.z, f1.w,
                       f2.x, f2.y, f2.z, f2.w, f3.x, f3.y, f3.z, f3.w};
        v16h av;
#pragma unroll
        for (int i = 0; i < 16; ++i) { av[i] = (_Float16)t[i]; sumsq += t[i] * t[i]; }
        a[s] = av;
    }
    // Row r split between lane r and lane r+16: combine halves.
    const float xsq = sumsq + __shfl_xor(sumsq, 16, 32);

    float minv[8];
    int   mini[8];
#pragma unroll
    for (int v = 0; v < 8; ++v) { minv[v] = 3.4e38f; mini[v] = 0; }

    for (int ch = 0; ch < K_CODES / CHUNK_CODES; ++ch) {
        __syncthreads();
        const _Float16* src = cb_h + (size_t)ch * CHUNK_CODES * D_DIM;
#ifdef USE_ASYNC_LDS
        // CDNA5 async DMA: global -> LDS without touching VGPRs (ASYNCcnt).
#pragma unroll
        for (int i = 0; i < 8; ++i) {
            int u = tid + i * 256;            // 2048 16-byte units
            int r = u >> 5, c16 = u & 31;
            __builtin_amdgcn_global_load_async_to_lds_b128(
                (AS1 v4i*)(src + r * D_DIM + c16 * 8),
                (AS3 v4i*)(&lds_cb[r * LDS_ROW_H + c16 * 8]),
                0, 0);
        }
        __builtin_amdgcn_s_wait_asynccnt(0);
#else
        // Fallback: stage through distinct registers so loads pipeline.
        v8h stage[8];
#pragma unroll
        for (int i = 0; i < 8; ++i) {
            int u = tid + i * 256;
            int r = u >> 5, c16 = u & 31;
            stage[i] = *(const v8h*)(src + r * D_DIM + c16 * 8);
        }
#pragma unroll
        for (int i = 0; i < 8; ++i) {
            int u = tid + i * 256;
            int r = u >> 5, c16 = u & 31;
            *(v8h*)(&lds_cb[r * LDS_ROW_H + c16 * 8]) = stage[i];
        }
#endif
        __syncthreads();

#pragma unroll
        for (int t = 0; t < 4; ++t) {
            const int codeLocal = t * 16 + l15;           // B column = code
            const int code      = ch * CHUNK_CODES + codeLocal;
            const float csq     = lds_csq[code];
            // B layout: lane holds column codeLocal, K in {0..7, 16..23} / {8..15, 24..31} per slice.
            const _Float16* bb = &lds_cb[codeLocal * LDS_ROW_H + hlf * 8];
            // Preload all B fragments so the 8 WMMAs issue back-to-back.
            v16h bfrag[8];
#pragma unroll
            for (int s = 0; s < 8; ++s) {
                v8h lo = *(const v8h*)(bb + s * 32);
                v8h hi = *(const v8h*)(bb + s * 32 + 16);
                bfrag[s] = __builtin_shufflevector(lo, hi, 0, 1, 2, 3, 4, 5, 6, 7,
                                                   8, 9, 10, 11, 12, 13, 14, 15);
            }
            v8f acc = {};
#pragma unroll
            for (int s = 0; s < 8; ++s) {
                acc = __builtin_amdgcn_wmma_f32_16x16x32_f16(
                    /*neg_a=*/false, a[s], /*neg_b=*/false, bfrag[s],
                    /*c_mod=*/(short)0, acc, /*reuse_a=*/false, /*reuse_b=*/false);
            }
            // acc[v] = dot(row rowBase+v(+8 for upper lanes), code). Distance sans ||x||^2.
#pragma unroll
            for (int v = 0; v < 8; ++v) {
                float d = csq - 2.0f * acc[v];
                if (d < minv[v]) { minv[v] = d; mini[v] = code; }  // ascending codes => first-min tiebreak
            }
        }
    }

    // Argmin across the 16 lanes (columns) of each half-wave.
#pragma unroll
    for (int off = 1; off < 16; off <<= 1) {
#pragma unroll
        for (int v = 0; v < 8; ++v) {
            float ov = __shfl_xor(minv[v], off, 32);
            int   oi = __shfl_xor(mini[v], off, 32);
            if (ov < minv[v] || (ov == minv[v] && oi < mini[v])) { minv[v] = ov; mini[v] = oi; }
        }
    }

    // Emit: gather winning code rows, straight-through output, indices, loss partial.
    float lossLocal = 0.f;
    for (int r = 0; r < 16; ++r) {
        const int srcLane = (r >> 3) << 4;                // 0 for rows 0-7, 16 for rows 8-15
        const int v       = r & 7;
        const int   bestIdx = __shfl(mini[v], srcLane, 32);
        const float bestVal = __shfl(minv[v], srcLane, 32);
        const float xs      = __shfl(xsq, r, 32);         // lane r holds x_sq of row rowBase+r
        const int orow = rowBase + r;
        const float* crow = cb + (size_t)bestIdx * D_DIM;
        const float* xr   = x + (size_t)orow * D_DIM;
        float*       op   = out + (size_t)orow * D_DIM;
#pragma unroll
        for (int i = 0; i < 2; ++i) {
            int c = lane * 8 + i * 4;
            float4 q  = *(const float4*)(crow + c);
            float4 xv = *(const float4*)(xr + c);
            float4 o;
            o.x = xv.x + (q.x - xv.x);
            o.y = xv.y + (q.y - xv.y);
            o.z = xv.z + (q.z - xv.z);
            o.w = xv.w + (q.w - xv.w);
            *(float4*)(op + c) = o;
        }
        if (lane == 0) {
            idx_out[orow] = (float)bestIdx;
            lossLocal += xs + bestVal;   // min dist = sum_d (x - q)^2 for this row
        }
    }
    if (lane == 0) partials[blockIdx.x * WAVES_PER_BLK + wave] = lossLocal;
}

// ---------------- Deterministic loss reduction ---------------------------------
__global__ __launch_bounds__(256) void vq_loss_finalize(const float* __restrict__ partials,
                                                        float* __restrict__ loss_out) {
    __shared__ float red[256];
    float s = 0.f;
    for (int i = threadIdx.x; i < (N_ROWS / 16); i += 256) s += partials[i];  // 4096 partials
    red[threadIdx.x] = s;
    __syncthreads();
    for (int stride = 128; stride > 0; stride >>= 1) {
        if (threadIdx.x < stride) red[threadIdx.x] += red[threadIdx.x + stride];
        __syncthreads();
    }
    if (threadIdx.x == 0)
        loss_out[0] = red[0] * (1.0f / 16777216.0f);   // COMMITMENT(=1) * mean over B*T*D
}

extern "C" void kernel_launch(void* const* d_in, const int* in_sizes, int n_in,
                              void* d_out, int out_size, void* d_ws, size_t ws_size,
                              hipStream_t stream) {
    const float* x  = (const float*)d_in[0];        // [16,4096,256] f32
    const float* cb = (const float*)d_in[1];        // [1024,256] f32

    float* out      = (float*)d_out;                                 // 16M f32
    float* idx_out  = out + (size_t)N_ROWS * D_DIM;                  // 65536 (as float)
    float* loss_out = idx_out + N_ROWS;                              // 1

    char* ws = (char*)d_ws;
    _Float16* cb_h   = (_Float16*)ws;                                         // 512 KB
    float*    c_sq   = (float*)(ws + (size_t)K_CODES * D_DIM * 2);            // 4 KB
    float*    parts  = (float*)(ws + (size_t)K_CODES * D_DIM * 2 + K_CODES*4);// 16 KB

    vq_prep_codebook<<<K_CODES / 8, 256, 0, stream>>>(cb, cb_h, c_sq);
    vq_main<<<N_ROWS / ROWS_PER_BLK, 256, 0, stream>>>(x, cb, cb_h, c_sq, out, idx_out, parts);
    vq_loss_finalize<<<1, 256, 0, stream>>>(parts, loss_out);
}
